// _ModConv_78365973282955
// MI455X (gfx1250) — matrix operand order repaired and checked by reading, not compiled, not run on hardware
//
#include <hip/hip_runtime.h>
#include <math.h>

// ---------------------------------------------------------------------------
// ModConv for MI455X (gfx1250, wave32, WMMA).
//   h  = silu(t@W1^T+b1); h = silu(h@W2^T+b2); s = h@W3^T+b3+1
//   bw = weight*s ; demod = rsqrt(sum_i bw^2 + eps) ; bw *= demod
//   out[b,o,l] = sum_i bw[b,o,i] * x[b,i,l]
// Main GEMM uses V_WMMA_F32_16X16X32_BF16 (bf16 operands, f32 accumulate):
// 34.4 GFLOP vs 268 MB DRAM floor (~11.5us @ 23.3 TB/s) -> memory bound.
// ---------------------------------------------------------------------------

#define B_ 16
#define I_ 512
#define O_ 512
#define T_ 256
#define L_ 4096

typedef __attribute__((ext_vector_type(16))) __bf16 v16bf;
typedef __attribute__((ext_vector_type(8)))  float  v8f;

__device__ __forceinline__ unsigned short f2bf(float f) {
  union { float f; unsigned u; } v; v.f = f;
  unsigned u = v.u;
  return (unsigned short)((u + 0x7fffu + ((u >> 16) & 1u)) >> 16); // RNE
}

__device__ __forceinline__ float silu(float x) { return x / (1.0f + expf(-x)); }

// --------------------------- Kernel 1: mod MLP -----------------------------
// One block per batch sample. h1/h2 live in LDS; writes s[B_, I_] (with +1).
__global__ __launch_bounds__(256) void modconv_mlp(
    const float* __restrict__ t,
    const float* __restrict__ W1, const float* __restrict__ b1,
    const float* __restrict__ W2, const float* __restrict__ b2,
    const float* __restrict__ W3, const float* __restrict__ b3,
    float* __restrict__ s)
{
  const int b   = blockIdx.x;
  const int tid = threadIdx.x;
  __shared__ float tin[T_], h1[T_], h2[T_];

  tin[tid] = t[b * T_ + tid];
  __syncthreads();

  float acc = b1[tid];
  const float* w = W1 + tid * T_;
  #pragma unroll 8
  for (int k = 0; k < T_; ++k) acc += w[k] * tin[k];
  h1[tid] = silu(acc);
  __syncthreads();

  acc = b2[tid];
  w = W2 + tid * T_;
  #pragma unroll 8
  for (int k = 0; k < T_; ++k) acc += w[k] * h1[k];
  h2[tid] = silu(acc);
  __syncthreads();

  #pragma unroll
  for (int rep = 0; rep < I_ / T_; ++rep) {
    const int i = tid + rep * T_;
    float a2 = b3[i] + 1.0f;
    const float* w3 = W3 + i * T_;
    #pragma unroll 8
    for (int k = 0; k < T_; ++k) a2 += w3[k] * h2[k];
    s[b * I_ + i] = a2;
  }
}

// --------------- Kernel 2: demodulate + swizzle A to bf16 ------------------
// One block per (b,o) row. Output layout = K-contiguous A-fragment order (k = 16*half + e):
//   element addr = ((b*O_+o)*16 + i/32)*32 + half*16 + e
//   half = bit4(i), e = i&15
// so in the GEMM each lane fetches its 16 bf16 A values with 2x b128 loads.
__global__ __launch_bounds__(256) void modconv_demod(
    const float* __restrict__ weight, const float* __restrict__ s,
    unsigned short* __restrict__ asw)
{
  const int bo  = blockIdx.x;
  const int b   = bo >> 9;        // / O_
  const int o   = bo & (O_ - 1);
  const int tid = threadIdx.x;
  __shared__ float red[256];

  const float2 wv = *(const float2*)(weight + (size_t)o * I_ + 2 * tid);
  const float2 sv = *(const float2*)(s + (size_t)b * I_ + 2 * tid);
  const float v0 = wv.x * sv.x;
  const float v1 = wv.y * sv.y;
  red[tid] = v0 * v0 + v1 * v1;
  __syncthreads();
  for (int st = 128; st > 0; st >>= 1) {
    if (tid < st) red[tid] += red[tid + st];
    __syncthreads();
  }
  const float dm = rsqrtf(red[0] + 1e-8f);

  const size_t rowBase = (size_t)(b * O_ + o) * (size_t)I_;
  #pragma unroll
  for (int j = 0; j < 2; ++j) {
    const int   i    = 2 * tid + j;
    const float val  = (j ? v1 : v0) * dm;
    const int   kb   = i >> 5;
    const int   r    = i & 31;
    const int   half = (r >> 4) & 1;
    const int   e    = r & 15;
    asw[rowBase + (size_t)kb * 32 + half * 16 + e] = f2bf(val);
  }
}

// ------------------------ Kernel 3: WMMA GEMM ------------------------------
// Per block: 128(O) x 256(L) tile of one batch. 8 waves = 2(M) x 4(N),
// each wave 64x64 via 4x4 tiles of v_wmma_f32_16x16x32_bf16.
// x slab (32 x 256 f32) staged in LDS as bf16, K-contiguous per column.
__global__ __launch_bounds__(256) void modconv_gemm(
    const float* __restrict__ x,
    const unsigned short* __restrict__ asw,
    float* __restrict__ out)
{
  const int b    = blockIdx.z;
  const int oBlk = blockIdx.y * 128;
  const int lBlk = blockIdx.x * 256;
  const int tid  = threadIdx.x;
  const int lane = tid & 31;
  const int wave = tid >> 5;
  const int wm   = wave >> 2;   // 0..1
  const int wn   = wave & 3;    // 0..3
  const int oW   = oBlk + wm * 64;
  const int lW   = wn * 64;     // column offset inside the 256-wide tile

  // column-major bf16 slab: column stride 40 ushorts (80B): 16B aligned
  // halves, and 80B/4 = 20-dword lane stride -> conflict-free over 16 lanes.
  __shared__ unsigned short ldsB[256 * 40];

  const float*          __restrict__ xb  = x   + (size_t)b * I_ * L_;
  const unsigned short* __restrict__ awb = asw + (size_t)b * O_ * I_;

  v8f acc[4][4] = {};

  const int convCol = (tid & 63) * 4;  // 4 consecutive L columns
  const int convK0  = tid >> 6;        // 0..3
  const int mrow    = lane & 15;
  const int halfSel = lane >> 4;

  for (int kb = 0; kb < 16; ++kb) {
    const int k0 = kb * 32;
    __syncthreads();   // previous iteration's frag reads done
    #pragma unroll
    for (int it = 0; it < 8; ++it) {
      const int k = it * 4 + convK0;   // 0..31
      const float4 v = *(const float4*)(xb + (size_t)(k0 + k) * L_ + lBlk + convCol);
      ldsB[(convCol + 0) * 40 + k] = f2bf(v.x);
      ldsB[(convCol + 1) * 40 + k] = f2bf(v.y);
      ldsB[(convCol + 2) * 40 + k] = f2bf(v.z);
      ldsB[(convCol + 3) * 40 + k] = f2bf(v.w);
    }
    __syncthreads();

    union Frag { v16bf v; uint4 q[2]; };
    Frag a[4], bq[4];

    #pragma unroll
    for (int mi = 0; mi < 4; ++mi) {
      // A frag straight from global (L2-resident, pre-swizzled)
      const size_t off = (((size_t)(oW + mi * 16 + mrow) * 16 + kb) * 32)
                         + (size_t)halfSel * 16;
      const uint4* p = (const uint4*)(awb + off);
      a[mi].q[0] = p[0];
      a[mi].q[1] = p[1];
    }
    #pragma unroll
    for (int ni = 0; ni < 4; ++ni) {
      const unsigned short* p = &ldsB[(lW + ni * 16 + mrow) * 40 + halfSel * 16];
      bq[ni].q[0] = ((const uint4*)p)[0];
      bq[ni].q[1] = ((const uint4*)p)[1];
    }

    #pragma unroll
    for (int mi = 0; mi < 4; ++mi)
      #pragma unroll
      for (int ni = 0; ni < 4; ++ni)
        acc[mi][ni] = __builtin_amdgcn_wmma_f32_16x16x32_bf16(
            false, a[mi].v, false, bq[ni].v,
            (short)0, acc[mi][ni], false, false);
  }

  // C/D layout: VGPR r -> lanes 0-15: (M=r, N=lane); lanes 16-31: (M=r+8).
  float* __restrict__ ob = out + (size_t)b * O_ * L_;
  const int rowAdd = halfSel * 8;
  #pragma unroll
  for (int mi = 0; mi < 4; ++mi) {
    #pragma unroll
    for (int ni = 0; ni < 4; ++ni) {
      const int col = lBlk + lW + ni * 16 + mrow;
      #pragma unroll
      for (int r = 0; r < 8; ++r)
        ob[(size_t)(oW + mi * 16 + rowAdd + r) * L_ + col] = acc[mi][ni][r];
    }
  }
}

// ---------------------------------------------------------------------------
extern "C" void kernel_launch(void* const* d_in, const int* in_sizes, int n_in,
                              void* d_out, int out_size, void* d_ws, size_t ws_size,
                              hipStream_t stream) {
  (void)in_sizes; (void)n_in; (void)out_size; (void)ws_size;
  const float* x      = (const float*)d_in[0];
  const float* t      = (const float*)d_in[1];
  const float* weight = (const float*)d_in[2];
  const float* W1     = (const float*)d_in[3];
  const float* b1     = (const float*)d_in[4];
  const float* W2     = (const float*)d_in[5];
  const float* b2     = (const float*)d_in[6];
  const float* W3     = (const float*)d_in[7];
  const float* b3     = (const float*)d_in[8];
  float* out = (float*)d_out;

  // workspace: s[B_*I_] f32 (32KB) then swizzled bf16 weights (8MB)
  float*          s   = (float*)d_ws;
  unsigned short* asw = (unsigned short*)((char*)d_ws + (size_t)B_ * I_ * sizeof(float));

  modconv_mlp<<<dim3(B_), dim3(256), 0, stream>>>(t, W1, b1, W2, b2, W3, b3, s);
  modconv_demod<<<dim3(B_ * O_), dim3(256), 0, stream>>>(weight, s, asw);
  modconv_gemm<<<dim3(L_ / 256, O_ / 128, B_), dim3(256), 0, stream>>>(x, asw, out);
}
